// MultiHeadAttention_90142773608874
// MI455X (gfx1250) — matrix operand order repaired and checked by reading, not compiled, not run on hardware
//
#include <hip/hip_runtime.h>
#include <hip/hip_bf16.h>
#include <stdint.h>

#define D_MODEL 768
#define NHEAD   12
#define DK      64
#define BATCH   2
#define SEQ     2048
#define MROWS   (BATCH*SEQ)   // 4096 rows for all projection GEMMs

typedef __attribute__((ext_vector_type(16))) __bf16 v16bf;
typedef __attribute__((ext_vector_type(8)))  __bf16 v8bf;
typedef __attribute__((ext_vector_type(8)))  float  v8f;

static __device__ __forceinline__ v16bf cat8(v8bf lo, v8bf hi) {
  return __builtin_shufflevector(lo, hi, 0,1,2,3,4,5,6,7,8,9,10,11,12,13,14,15);
}

// A-fragment for v_wmma_f32_16x16x32_bf16 from a row-major tile.
// `row` points at element [lane&15][k0] of the 16x32 tile.
// Lane layout: lanes 0-15 hold K 0-15, lanes 16-31 hold K 16-31 (matches B loads).
static __device__ __forceinline__ v16bf load_a_frag(const __bf16* row, int laneHi) {
  return *(const v16bf*)(row + 16 * laneHi);
}

// ---------------------------------------------------------------------------
// fp32 -> bf16 conversion
// ---------------------------------------------------------------------------
__global__ void cvt_f32_bf16(const float* __restrict__ in, __bf16* __restrict__ out, int n) {
  int i = blockIdx.x * blockDim.x + threadIdx.x;
  if (i < n) out[i] = (__bf16)in[i];
}

// ---------------------------------------------------------------------------
// Y[M,N] = X[M,K] @ W[N,K]^T + bias   (M=4096, N=K=768)
// mode 0: fp32 row-major out; mode 1: bf16 row-major out;
// mode 2: bf16 transposed-per-head out  Vt[b][h][d][s]  (for the P@V operand)
// One wave computes a 32x64 output strip (8 accumulator tiles), with explicit
// register double-buffering of the next K-step's fragments so loads overlap
// the WMMAs instead of serializing on s_wait_loadcnt 0.
// ---------------------------------------------------------------------------
__global__ void gemm_bias_kernel(const __bf16* __restrict__ X,
                                 const __bf16* __restrict__ W,
                                 const float* __restrict__ bias,
                                 void* __restrict__ out, int mode) {
  const int lane   = threadIdx.x & 31;
  const int laneHi = lane >> 4;
  const int lm     = lane & 15;
  const int wave   = blockIdx.x * (blockDim.x >> 5) + (threadIdx.x >> 5);
  const int NG     = D_MODEL / 64;          // 12 N-groups
  const int mt     = wave / NG;             // 0..127
  const int ng     = wave % NG;
  if (mt >= MROWS / 32) return;
  const int m0 = mt * 32;

  v8f acc[2][4] = {};
  const __bf16* xrow0 = X + (size_t)(m0 + lm) * D_MODEL;
  const __bf16* xrow1 = X + (size_t)(m0 + 16 + lm) * D_MODEL;
  const __bf16* wrow  = W + (size_t)(ng * 64 + lm) * D_MODEL + 16 * laneHi;

  // prologue: fragments for k0 = 0
  v16bf a0c = load_a_frag(xrow0, laneHi);
  v16bf a1c = load_a_frag(xrow1, laneHi);
  v16bf bc0 = *(const v16bf*)(wrow + 0 * 16 * D_MODEL);
  v16bf bc1 = *(const v16bf*)(wrow + 1 * 16 * D_MODEL);
  v16bf bc2 = *(const v16bf*)(wrow + 2 * 16 * D_MODEL);
  v16bf bc3 = *(const v16bf*)(wrow + 3 * 16 * D_MODEL);

#pragma unroll
  for (int k0 = 0; k0 < D_MODEL; k0 += 32) {
    v16bf a0n = a0c, a1n = a1c, bn0 = bc0, bn1 = bc1, bn2 = bc2, bn3 = bc3;
    if (k0 + 32 < D_MODEL) {                 // issue next-step loads first
      const int kn = k0 + 32;
      a0n = load_a_frag(xrow0 + kn, laneHi);
      a1n = load_a_frag(xrow1 + kn, laneHi);
      bn0 = *(const v16bf*)(wrow + kn + 0 * 16 * D_MODEL);
      bn1 = *(const v16bf*)(wrow + kn + 1 * 16 * D_MODEL);
      bn2 = *(const v16bf*)(wrow + kn + 2 * 16 * D_MODEL);
      bn3 = *(const v16bf*)(wrow + kn + 3 * 16 * D_MODEL);
    }
    acc[0][0] = __builtin_amdgcn_wmma_f32_16x16x32_bf16(false, a0c, false, bc0, (short)0, acc[0][0], false, false);
    acc[0][1] = __builtin_amdgcn_wmma_f32_16x16x32_bf16(false, a0c, false, bc1, (short)0, acc[0][1], false, false);
    acc[0][2] = __builtin_amdgcn_wmma_f32_16x16x32_bf16(false, a0c, false, bc2, (short)0, acc[0][2], false, false);
    acc[0][3] = __builtin_amdgcn_wmma_f32_16x16x32_bf16(false, a0c, false, bc3, (short)0, acc[0][3], false, false);
    acc[1][0] = __builtin_amdgcn_wmma_f32_16x16x32_bf16(false, a1c, false, bc0, (short)0, acc[1][0], false, false);
    acc[1][1] = __builtin_amdgcn_wmma_f32_16x16x32_bf16(false, a1c, false, bc1, (short)0, acc[1][1], false, false);
    acc[1][2] = __builtin_amdgcn_wmma_f32_16x16x32_bf16(false, a1c, false, bc2, (short)0, acc[1][2], false, false);
    acc[1][3] = __builtin_amdgcn_wmma_f32_16x16x32_bf16(false, a1c, false, bc3, (short)0, acc[1][3], false, false);
    a0c = a0n; a1c = a1n; bc0 = bn0; bc1 = bn1; bc2 = bn2; bc3 = bn3;
  }

#pragma unroll
  for (int mr = 0; mr < 2; ++mr) {
#pragma unroll
    for (int nt = 0; nt < 4; ++nt) {
      const int j  = ng * 64 + nt * 16 + lm;
      const float bj = bias[j];
      if (mode == 0) {
        float* o = (float*)out;
#pragma unroll
        for (int i = 0; i < 8; ++i)
          o[(size_t)(m0 + mr * 16 + i + 8 * laneHi) * D_MODEL + j] = acc[mr][nt][i] + bj;
      } else if (mode == 1) {
        __bf16* o = (__bf16*)out;
#pragma unroll
        for (int i = 0; i < 8; ++i)
          o[(size_t)(m0 + mr * 16 + i + 8 * laneHi) * D_MODEL + j] = (__bf16)(acc[mr][nt][i] + bj);
      } else {
        // Vt[b][h][d][s]
        __bf16* o = (__bf16*)out;
        const int h = j >> 6, d = j & 63;
#pragma unroll
        for (int i = 0; i < 8; ++i) {
          const int sg = m0 + mr * 16 + i + 8 * laneHi;   // 0..4095
          const int b = sg >> 11, srow = sg & (SEQ - 1);
          o[((size_t)(b * NHEAD + h) * DK + d) * SEQ + srow] = (__bf16)(acc[mr][nt][i] + bj);
        }
      }
    }
  }
}

// ---------------------------------------------------------------------------
// Attention, recompute-not-spill: one wave per (b, h, 16 query rows).
// Pass 1: WMMA scores -> exact per-row softmax stats (online max/sum), no
//         global writes at all.
// Pass 2: recompute scores (cheap vs HBM), write FINAL normalized attn
//         probabilities once (only attn traffic = one 400MB write), and
//         accumulate x = P @ V via LDS-relaid bf16 P fragments.
// ---------------------------------------------------------------------------
__global__ void attn_kernel(const __bf16* __restrict__ Qp,
                            const __bf16* __restrict__ Kp,
                            const __bf16* __restrict__ Vt,
                            float* __restrict__ attn,
                            __bf16* __restrict__ xb) {
  __shared__ __bf16 plds[4][16 * 32];   // per-wave 16x32 P tile
  const int wv     = threadIdx.x >> 5;
  const int lane   = threadIdx.x & 31;
  const int laneHi = lane >> 4;
  const int lm     = lane & 15;
  const int wave   = blockIdx.x * 4 + wv;
  if (wave >= BATCH * NHEAD * (SEQ / 16)) return;

  const int mt = wave & (SEQ / 16 - 1);
  const int bh = wave >> 7;               // SEQ/16 == 128
  const int h  = bh % NHEAD;
  const int b  = bh / NHEAD;
  const int s0 = mt * 16;

  const __bf16* qrow = Qp + (size_t)(b * SEQ + s0 + lm) * D_MODEL + h * DK;
  const v16bf qa0 = load_a_frag(qrow, laneHi);
  const v16bf qa1 = load_a_frag(qrow + 32, laneHi);

  const __bf16* khead = Kp + (size_t)(b * SEQ) * D_MODEL + h * DK;
  const __bf16* vhead = Vt + (size_t)(b * NHEAD + h) * DK * SEQ;
  float* attnBase = attn + ((size_t)(b * NHEAD + h) * SEQ + s0) * SEQ;

  float rmax[8], rsum[8];
#pragma unroll
  for (int i = 0; i < 8; ++i) { rmax[i] = -1e30f; rsum[i] = 0.0f; }

  // ---------------- pass 1: stats only ----------------
  for (int nn = 0; nn < SEQ; nn += 32) {
    const __bf16* k0row = khead + (size_t)(nn + lm) * D_MODEL + 16 * laneHi;
    const __bf16* k1row = k0row + (size_t)16 * D_MODEL;
    if (nn + 32 < SEQ)
      __builtin_prefetch(k0row + (size_t)32 * D_MODEL, 0, 0);
    v16bf kb00 = *(const v16bf*)(k0row);
    v16bf kb01 = *(const v16bf*)(k0row + 32);
    v16bf kb10 = *(const v16bf*)(k1row);
    v16bf kb11 = *(const v16bf*)(k1row + 32);

    v8f c0 = {}, c1 = {};
    c0 = __builtin_amdgcn_wmma_f32_16x16x32_bf16(false, qa0, false, kb00, (short)0, c0, false, false);
    c0 = __builtin_amdgcn_wmma_f32_16x16x32_bf16(false, qa1, false, kb01, (short)0, c0, false, false);
    c1 = __builtin_amdgcn_wmma_f32_16x16x32_bf16(false, qa0, false, kb10, (short)0, c1, false, false);
    c1 = __builtin_amdgcn_wmma_f32_16x16x32_bf16(false, qa1, false, kb11, (short)0, c1, false, false);

    float tm[8];
#pragma unroll
    for (int i = 0; i < 8; ++i) tm[i] = fmaxf(c0[i], c1[i]) * 0.125f;
#pragma unroll
    for (int m = 8; m >= 1; m >>= 1)
#pragma unroll
      for (int i = 0; i < 8; ++i) tm[i] = fmaxf(tm[i], __shfl_xor(tm[i], m, 16));

    float nm[8], al[8], ps[8];
#pragma unroll
    for (int i = 0; i < 8; ++i) {
      nm[i] = fmaxf(rmax[i], tm[i]);
      al[i] = __expf(rmax[i] - nm[i]);
      rmax[i] = nm[i];
      ps[i] = __expf(c0[i] * 0.125f - nm[i]) + __expf(c1[i] * 0.125f - nm[i]);
    }
#pragma unroll
    for (int m = 8; m >= 1; m >>= 1)
#pragma unroll
      for (int i = 0; i < 8; ++i) ps[i] += __shfl_xor(ps[i], m, 16);
#pragma unroll
    for (int i = 0; i < 8; ++i) rsum[i] = rsum[i] * al[i] + ps[i];
  }

  float inv[8];
#pragma unroll
  for (int i = 0; i < 8; ++i) inv[i] = 1.0f / rsum[i];

  // ---------------- pass 2: recompute, write p, accumulate x ----------------
  v8f xacc[4] = {};
  for (int nn = 0; nn < SEQ; nn += 32) {
    const __bf16* k0row = khead + (size_t)(nn + lm) * D_MODEL + 16 * laneHi;
    const __bf16* k1row = k0row + (size_t)16 * D_MODEL;
    v16bf kb00 = *(const v16bf*)(k0row);
    v16bf kb01 = *(const v16bf*)(k0row + 32);
    v16bf kb10 = *(const v16bf*)(k1row);
    v16bf kb11 = *(const v16bf*)(k1row + 32);

    v8f c0 = {}, c1 = {};
    c0 = __builtin_amdgcn_wmma_f32_16x16x32_bf16(false, qa0, false, kb00, (short)0, c0, false, false);
    c0 = __builtin_amdgcn_wmma_f32_16x16x32_bf16(false, qa1, false, kb01, (short)0, c0, false, false);
    c1 = __builtin_amdgcn_wmma_f32_16x16x32_bf16(false, qa0, false, kb10, (short)0, c1, false, false);
    c1 = __builtin_amdgcn_wmma_f32_16x16x32_bf16(false, qa1, false, kb11, (short)0, c1, false, false);

    // final normalized probabilities; the ONLY write to the attn tensor
#pragma unroll
    for (int i = 0; i < 8; ++i) {
      const float p0 = __expf(c0[i] * 0.125f - rmax[i]) * inv[i];
      const float p1 = __expf(c1[i] * 0.125f - rmax[i]) * inv[i];
      attnBase[(size_t)(i + 8 * laneHi) * SEQ + nn + lm]      = p0;
      attnBase[(size_t)(i + 8 * laneHi) * SEQ + nn + 16 + lm] = p1;
      plds[wv][(i + 8 * laneHi) * 32 + lm]      = (__bf16)p0;
      plds[wv][(i + 8 * laneHi) * 32 + 16 + lm] = (__bf16)p1;
    }
    asm volatile("s_wait_dscnt 0" ::: "memory");   // wave-wide P tile visible

    const v16bf pa = load_a_frag(&plds[wv][lm * 32], laneHi);
    const __bf16* vrow = vhead + (size_t)lm * SEQ + nn + 16 * laneHi;
    if (nn + 32 < SEQ)
      __builtin_prefetch(vrow + 32, 0, 0);
#pragma unroll
    for (int dt = 0; dt < 4; ++dt) {
      v16bf vbf = *(const v16bf*)(vrow + (size_t)(dt * 16) * SEQ);
      xacc[dt] = __builtin_amdgcn_wmma_f32_16x16x32_bf16(
          false, pa, false, vbf, (short)0, xacc[dt], false, false);
    }
  }

#pragma unroll
  for (int dt = 0; dt < 4; ++dt)
#pragma unroll
    for (int i = 0; i < 8; ++i)
      xb[(size_t)(b * SEQ + s0 + i + 8 * laneHi) * D_MODEL + h * DK + dt * 16 + lm] =
          (__bf16)xacc[dt][i];
}

// ---------------------------------------------------------------------------
extern "C" void kernel_launch(void* const* d_in, const int* in_sizes, int n_in,
                              void* d_out, int out_size, void* d_ws, size_t ws_size,
                              hipStream_t stream) {
  const float* q   = (const float*)d_in[0];
  const float* k   = (const float*)d_in[1];
  const float* v   = (const float*)d_in[2];
  const float* w_q = (const float*)d_in[3];
  const float* b_q = (const float*)d_in[4];
  const float* w_k = (const float*)d_in[5];
  const float* b_k = (const float*)d_in[6];
  const float* w_v = (const float*)d_in[7];
  const float* b_v = (const float*)d_in[8];
  const float* w_o = (const float*)d_in[9];
  const float* b_o = (const float*)d_in[10];

  const size_t NACT = (size_t)MROWS * D_MODEL;    // 3,145,728
  const size_t NWGT = (size_t)D_MODEL * D_MODEL;  //   589,824
  float* out  = (float*)d_out;
  float* attn = out + NACT;                       // tuple: (out, attn)

  // workspace partition (bf16 = 2 bytes)
  char* ws = (char*)d_ws;
  size_t off = 0;
  auto alloc = [&](size_t bytes) { char* p = ws + off; off += (bytes + 255) & ~(size_t)255; return p; };
  __bf16* qb   = (__bf16*)alloc(NACT * 2);
  __bf16* kb   = (__bf16*)alloc(NACT * 2);
  __bf16* vb   = (__bf16*)alloc(NACT * 2);
  __bf16* wqb  = (__bf16*)alloc(NWGT * 2);
  __bf16* wkb  = (__bf16*)alloc(NWGT * 2);
  __bf16* wvb  = (__bf16*)alloc(NWGT * 2);
  __bf16* wob  = (__bf16*)alloc(NWGT * 2);
  __bf16* Qp   = (__bf16*)alloc(NACT * 2);
  __bf16* Kp   = (__bf16*)alloc(NACT * 2);
  __bf16* Vt   = (__bf16*)alloc(NACT * 2);
  __bf16* xb   = (__bf16*)alloc(NACT * 2);

  // 1) fp32 -> bf16 conversions
  const int CT = 256;
  cvt_f32_bf16<<<(NACT + CT - 1) / CT, CT, 0, stream>>>(q, qb, (int)NACT);
  cvt_f32_bf16<<<(NACT + CT - 1) / CT, CT, 0, stream>>>(k, kb, (int)NACT);
  cvt_f32_bf16<<<(NACT + CT - 1) / CT, CT, 0, stream>>>(v, vb, (int)NACT);
  cvt_f32_bf16<<<(NWGT + CT - 1) / CT, CT, 0, stream>>>(w_q, wqb, (int)NWGT);
  cvt_f32_bf16<<<(NWGT + CT - 1) / CT, CT, 0, stream>>>(w_k, wkb, (int)NWGT);
  cvt_f32_bf16<<<(NWGT + CT - 1) / CT, CT, 0, stream>>>(w_v, wvb, (int)NWGT);
  cvt_f32_bf16<<<(NWGT + CT - 1) / CT, CT, 0, stream>>>(w_o, wob, (int)NWGT);

  // 2) QKV projections (WMMA GEMMs). 1536 waves (32x64 strips), 8/block.
  const int GB = (MROWS / 32) * (D_MODEL / 64) / 8;  // 192
  gemm_bias_kernel<<<GB, 256, 0, stream>>>(qb, wqb, b_q, Qp, 1);
  gemm_bias_kernel<<<GB, 256, 0, stream>>>(kb, wkb, b_k, Kp, 1);
  gemm_bias_kernel<<<GB, 256, 0, stream>>>(vb, wvb, b_v, Vt, 2);

  // 3) attention: stats pass + recompute/write/PV pass
  const int AW = BATCH * NHEAD * (SEQ / 16);         // 3072 waves
  attn_kernel<<<AW / 4, 128, 0, stream>>>(Qp, Kp, Vt, attn, xb);

  // 4) output projection -> fp32 out
  gemm_bias_kernel<<<GB, 256, 0, stream>>>(xb, wob, b_o, out, 0);

  (void)in_sizes; (void)n_in; (void)out_size; (void)ws_size;
}
